// HeterogeneousTemporalHypergraphNN_50689204027485
// MI455X (gfx1250) — matrix operand order repaired and checked by reading, not compiled, not run on hardware
//
#include <hip/hip_runtime.h>
#include <hip/hip_bf16.h>

// Problem constants (from reference)
#define NN 100000   // nodes
#define NE 1024     // hyperedges
#define NC 128      // channels
#define KCHUNKS 64  // split-K partials for edge aggregation
#define KB 1568     // nodes per K-chunk (64 * 1568 = 100352 >= NN, padded)

typedef _Float16 v16h __attribute__((ext_vector_type(16)));
typedef float    v8f  __attribute__((ext_vector_type(8)));
typedef float    v4f  __attribute__((ext_vector_type(4)));
typedef int      v4i  __attribute__((ext_vector_type(4)));

#if __has_builtin(__builtin_amdgcn_global_load_async_to_lds_b128)
#define HAVE_ASYNC_LDS 1
typedef __attribute__((address_space(1))) v4i* g4ptr_t;   // global (AS1) v4i*
typedef __attribute__((address_space(3))) v4i* l4ptr_t;   // LDS (AS3) v4i*
#endif

__device__ __forceinline__ void wait_asynccnt0() {
#if __has_builtin(__builtin_amdgcn_s_wait_asynccnt)
  __builtin_amdgcn_s_wait_asynccnt(0);
#else
  asm volatile("s_wait_asynccnt 0x0" ::: "memory");
#endif
}

__device__ __forceinline__ v8f wmma_f16(v16h a, v16h b, v8f c) {
  // D = A(16x32 f16) * B(32x16 f16) + C(16x16 f32)
  return __builtin_amdgcn_wmma_f32_16x16x32_f16(
      /*neg_a=*/false, a, /*neg_b=*/false, b,
      /*c_mod=*/(short)0, c, /*reuse_a=*/false, /*reuse_b=*/false);
}

__device__ __forceinline__ void atomicMaxFloatBits(unsigned* p, float v) {
  atomicMax(p, __float_as_uint(v));   // valid for non-negative floats
}

// ---------------------------------------------------------------------------
// K0: init workspace scalars (maxdeg bits, maxsens bits)
// ---------------------------------------------------------------------------
__global__ void init_scal_kernel(unsigned* scal) {
  if (threadIdx.x < 2) scal[threadIdx.x] = 0u;
}

// ---------------------------------------------------------------------------
// K1: he_part[chunk] = (incidence^T @ X) over this chunk's K-range.
// grid = (NE/128, KCHUNKS), block = 256 (8 waves).
// Block tile: 128 edges x 128 channels; wave w owns C-tile w, 8 E-subtiles.
// Double-buffered LDS, register prefetch, one barrier per k-step.
// ---------------------------------------------------------------------------
__device__ __forceinline__ void k1_load(const float* __restrict__ X,
                                        const float* __restrict__ inc,
                                        long kbase, int tid, int e0,
                                        v4f* bv, v4f* av) {
  if (kbase + 32 <= (long)NN) {          // uniform fast path: no clamps/masks
#pragma unroll
    for (int i = 0; i < 4; ++i) {
      int idx = tid + i * 256;           // 0..1023
      int kr  = idx >> 5;                // 0..31
      int c4  = (idx & 31) << 2;         // 0..124
      bv[i] = *(const v4f*)(X + (kbase + kr) * NC + c4);
      av[i] = __builtin_nontemporal_load(
          (const v4f*)(inc + (kbase + kr) * (long)NE + e0 + c4));
    }
  } else {                               // rare padded tail
#pragma unroll
    for (int i = 0; i < 4; ++i) {
      int idx = tid + i * 256;
      int kr  = idx >> 5;
      int c4  = (idx & 31) << 2;
      long n  = kbase + kr;
      long ncl = (n < NN) ? n : (long)(NN - 1);
      float m = (n < NN) ? 1.0f : 0.0f;
      bv[i] = *(const v4f*)(X + ncl * NC + c4) * m;
      av[i] = __builtin_nontemporal_load(
                  (const v4f*)(inc + ncl * (long)NE + e0 + c4)) * m;
    }
  }
}

__global__ __launch_bounds__(256)
void edge_agg_kernel(const float* __restrict__ inc,   // (NN, NE)
                     const float* __restrict__ X,     // (NN, NC)
                     float* __restrict__ part) {      // (KCHUNKS, NE, NC)
  __shared__ _Float16 AT[2][128][40];    // [buf][e][k]  : inc^T tile (f16)
  __shared__ _Float16 BT[2][128][40];    // [buf][c][k]  : X tile transposed

  const int tid  = threadIdx.x;
  const int lane = tid & 31;
  const int w    = tid >> 5;
  const int hi   = lane >> 4;
  const int lm   = lane & 15;
  const int e0   = blockIdx.x * 128;
  const long kchunk0 = (long)blockIdx.y * KB;

  v8f acc[8] = {};
  v4f bv[4], av[4];
  k1_load(X, inc, kchunk0, tid, e0, bv, av);

  int p = 0;
  for (int kk = 0; kk < KB; kk += 32) {
    // ---- commit prefetched tile to LDS buffer p ----
#pragma unroll
    for (int i = 0; i < 4; ++i) {
      int idx = tid + i * 256;
      int kr  = idx >> 5;
      int c4  = (idx & 31) << 2;
#pragma unroll
      for (int q = 0; q < 4; ++q) {
        BT[p][c4 + q][kr] = (_Float16)bv[i][q];
        AT[p][c4 + q][kr] = (_Float16)av[i][q];
      }
    }
    __syncthreads();
    // ---- prefetch next tile into registers (overlaps WMMA below) ----
    if (kk + 32 < KB) k1_load(X, inc, kchunk0 + kk + 32, tid, e0, bv, av);

    // ---- compute: 8 WMMA per wave ----
    const int cc = w * 16 + lm;
    v16h bfrag;
#pragma unroll
    for (int j = 0; j < 16; ++j) bfrag[j] = BT[p][cc][hi * 16 + j];
#pragma unroll
    for (int r = 0; r < 8; ++r) {
      v16h afrag;
#pragma unroll
      for (int j = 0; j < 16; ++j) {
        int ka = hi * 16 + j;
        afrag[j] = AT[p][r * 16 + lm][ka];
      }
      acc[r] = wmma_f16(afrag, bfrag, acc[r]);
    }
    p ^= 1;
  }

  const int cc = w * 16 + lm;
#pragma unroll
  for (int r = 0; r < 8; ++r)
#pragma unroll
    for (int i = 0; i < 8; ++i) {
      int e = e0 + r * 16 + i + 8 * hi;
      part[(size_t)blockIdx.y * (NE * NC) + (size_t)e * NC + cc] = acc[r][i];
    }
}

// ---------------------------------------------------------------------------
// K2: he_pre = sum over partials (deterministic reduction)
// ---------------------------------------------------------------------------
__global__ void he_reduce_kernel(const float* __restrict__ part,
                                 float* __restrict__ he_pre) {
  int idx = blockIdx.x * 256 + threadIdx.x;      // < NE*NC
  float s = 0.0f;
#pragma unroll 8
  for (int p = 0; p < KCHUNKS; ++p) s += part[(size_t)p * (NE * NC) + idx];
  he_pre[idx] = s;
}

// ---------------------------------------------------------------------------
// K3: softmax(edge_attention) -> attn, plus Shannon entropy. One block.
// ---------------------------------------------------------------------------
__global__ void attn_entropy_kernel(const float* __restrict__ ea,
                                    float* __restrict__ out_attn,
                                    float* __restrict__ out_ent) {
  __shared__ float red[256];
  __shared__ float sh;
  const int t = threadIdx.x;

  float m = -3.4e38f;
  for (int e = t; e < NE; e += 256) m = fmaxf(m, ea[e]);
  red[t] = m; __syncthreads();
  for (int s = 128; s > 0; s >>= 1) { if (t < s) red[t] = fmaxf(red[t], red[t + s]); __syncthreads(); }
  if (t == 0) sh = red[0];
  __syncthreads();
  const float gm = sh;

  float sum = 0.0f;
  for (int e = t; e < NE; e += 256) sum += expf(ea[e] - gm);
  red[t] = sum; __syncthreads();
  for (int s = 128; s > 0; s >>= 1) { if (t < s) red[t] += red[t + s]; __syncthreads(); }
  if (t == 0) sh = red[0];
  __syncthreads();
  const float S = sh;

  float s2 = 0.0f;
  for (int e = t; e < NE; e += 256) { float a = expf(ea[e] - gm) / S; out_attn[e] = a; s2 += a; }
  red[t] = s2; __syncthreads();
  for (int s = 128; s > 0; s >>= 1) { if (t < s) red[t] += red[t + s]; __syncthreads(); }
  if (t == 0) sh = red[0];
  __syncthreads();
  const float S2 = sh + 1e-8f;

  float ent = 0.0f;
  for (int e = t; e < NE; e += 256) { float p = out_attn[e] / S2; ent -= p * logf(p + 1e-8f); }
  red[t] = ent; __syncthreads();
  for (int s = 128; s > 0; s >>= 1) { if (t < s) red[t] += red[t + s]; __syncthreads(); }
  if (t == 0) out_ent[0] = red[0];
}

// ---------------------------------------------------------------------------
// K4: he = (he_pre @ W1 + b1) * attn[e]   (exact fp32, tiny GEMM)
// ---------------------------------------------------------------------------
__global__ void he_linear_kernel(const float* __restrict__ he_pre,
                                 const float* __restrict__ W1,
                                 const float* __restrict__ b1,
                                 const float* __restrict__ attn,
                                 float* __restrict__ out_he) {
  int idx = blockIdx.x * 256 + threadIdx.x;      // < NE*NC
  int e = idx >> 7, c = idx & (NC - 1);
  float s = b1[c];
  for (int k = 0; k < NC; ++k) s += he_pre[e * NC + k] * W1[k * NC + c];
  out_he[idx] = s * attn[e];
}

// ---------------------------------------------------------------------------
// K5: node_out = ((inc @ he) @ W2 + b2) + X ; also deg[n] = row-sum(inc).
// grid = NN/32 blocks, 256 threads (8 waves); wave w owns C-tile w, 2 row-
// subtiles. Incidence tile async-prefetched global->LDS (double-buffered).
// ---------------------------------------------------------------------------
__device__ __forceinline__ void issue_asyncA(const float* __restrict__ inc,
                                             long n0, int k0, int tid,
                                             float (*AF)[36]) {
  int r  = tid >> 3;                 // 0..31
  int c4 = (tid & 7) << 2;           // 0..28
  const float* gp = inc + (n0 + r) * (long)NE + k0 + c4;
#ifdef HAVE_ASYNC_LDS
  __builtin_amdgcn_global_load_async_to_lds_b128((g4ptr_t)gp,
                                                 (l4ptr_t)&AF[r][c4], 0, 0);
#else
  *(v4f*)&AF[r][c4] = __builtin_nontemporal_load((const v4f*)gp);
#endif
}

__global__ __launch_bounds__(256)
void propagate_kernel(const float* __restrict__ inc,    // (NN, NE)
                      const float* __restrict__ he,     // (NE, NC)
                      const float* __restrict__ W2,     // (NC, NC)
                      const float* __restrict__ b2,     // (NC,)
                      const float* __restrict__ X,      // (NN, NC)
                      float* __restrict__ out_node,     // (NN, NC)
                      float* __restrict__ degw,         // (NN,)
                      unsigned* __restrict__ maxdeg_bits) {
  __shared__ float    AF[2][32][36];   // [buf][r][kr], incidence tile (fp32)
  __shared__ _Float16 BT[2][128][40];  // [buf][c][kr], he chunk transposed
  __shared__ _Float16 Tmp[32][136];    // phase-1 result (32 x NC, f16)

  const int tid  = threadIdx.x;
  const int lane = tid & 31;
  const int w    = tid >> 5;
  const int hi   = lane >> 4;
  const int lm   = lane & 15;
  const long n0  = (long)blockIdx.x * 32;
  const int cc   = w * 16 + lm;

  v8f acc[2] = {};
  float degacc = 0.0f;
  v4f bv[4];

  // prologue: async A for step 0 + he regs for step 0
  issue_asyncA(inc, n0, 0, tid, AF[0]);
#pragma unroll
  for (int i = 0; i < 4; ++i) {
    int idx = tid + i * 256;
    int kr  = idx >> 5;
    int c4  = (idx & 31) << 2;
    bv[i] = *(const v4f*)(he + kr * NC + c4);
  }

  int p = 0;
  for (int k0 = 0; k0 < NE; k0 += 32) {
    // commit he regs to LDS buffer p
#pragma unroll
    for (int i = 0; i < 4; ++i) {
      int idx = tid + i * 256;
      int kr  = idx >> 5;
      int c4  = (idx & 31) << 2;
#pragma unroll
      for (int q = 0; q < 4; ++q) BT[p][c4 + q][kr] = (_Float16)bv[i][q];
    }
    wait_asynccnt0();                // A tile for this step has landed
    __syncthreads();

    // prefetch next step: async A + he regs (overlap with WMMA below)
    if (k0 + 32 < NE) {
      issue_asyncA(inc, n0, k0 + 32, tid, AF[p ^ 1]);
#pragma unroll
      for (int i = 0; i < 4; ++i) {
        int idx = tid + i * 256;
        int kr  = idx >> 5;
        int c4  = (idx & 31) << 2;
        bv[i] = *(const v4f*)(he + (k0 + 32 + kr) * NC + c4);
      }
    }

    // compute: 2 WMMA per wave + degree fold
    v16h bfrag;
#pragma unroll
    for (int j = 0; j < 16; ++j) bfrag[j] = BT[p][cc][hi * 16 + j];
#pragma unroll
    for (int r = 0; r < 2; ++r) {
      v16h afrag;
#pragma unroll
      for (int j = 0; j < 16; ++j) {
        int ka = hi * 16 + j;
        afrag[j] = (_Float16)AF[p][r * 16 + lm][ka];
      }
      acc[r] = wmma_f16(afrag, bfrag, acc[r]);
    }
    if (tid < 32) {                  // exact fp32 degree row sums
#pragma unroll
      for (int kr = 0; kr < 32; ++kr) degacc += AF[p][tid][kr];
    }
    p ^= 1;
  }

  if (tid < 32) {
    degw[n0 + tid] = degacc;
    atomicMaxFloatBits(maxdeg_bits, degacc);
  }

  // stash phase-1 tiles as f16 for the W2 GEMM
  __syncthreads();
#pragma unroll
  for (int r = 0; r < 2; ++r)
#pragma unroll
    for (int i = 0; i < 8; ++i) Tmp[r * 16 + i + 8 * hi][cc] = (_Float16)acc[r][i];
  __syncthreads();

  // phase 2: (32 x NC) @ W2 via WMMA, K = 128; B fragments from L2-hot W2
  v8f acc2[2] = {};
#pragma unroll
  for (int k0 = 0; k0 < NC; k0 += 32) {
    v16h bfrag;
#pragma unroll
    for (int j = 0; j < 16; ++j)
      bfrag[j] = (_Float16)W2[(k0 + hi * 16 + j) * NC + cc];
#pragma unroll
    for (int r = 0; r < 2; ++r) {
      v16h afrag;
#pragma unroll
      for (int j = 0; j < 16; ++j) {
        int ka = hi * 16 + j;
        afrag[j] = Tmp[r * 16 + lm][k0 + ka];
      }
      acc2[r] = wmma_f16(afrag, bfrag, acc2[r]);
    }
  }

#pragma unroll
  for (int r = 0; r < 2; ++r)
#pragma unroll
    for (int i = 0; i < 8; ++i) {
      long n = n0 + r * 16 + i + 8 * hi;
      out_node[n * NC + cc] = acc2[r][i] + b2[cc] + X[n * NC + cc];
    }
}

// ---------------------------------------------------------------------------
// K6: raw sensitivity = ||X_row|| * deg/(maxdeg+eps); track max
// ---------------------------------------------------------------------------
__global__ void sens_raw_kernel(const float* __restrict__ X,
                                const float* __restrict__ degw,
                                const unsigned* __restrict__ maxdeg_bits,
                                float* __restrict__ raw,
                                unsigned* __restrict__ maxsens_bits) {
  __shared__ float red[128];
  const long n = blockIdx.x;
  const int t = threadIdx.x;
  float v = X[n * NC + t];
  red[t] = v * v; __syncthreads();
  for (int s = 64; s > 0; s >>= 1) { if (t < s) red[t] += red[t + s]; __syncthreads(); }
  if (t == 0) {
    float norm = sqrtf(red[0]);
    float md = __uint_as_float(*maxdeg_bits);
    float sv = norm * (degw[n] / (md + 1e-8f));
    raw[n] = sv;
    atomicMaxFloatBits(maxsens_bits, sv);
  }
}

__global__ void sens_norm_kernel(const float* __restrict__ raw,
                                 const unsigned* __restrict__ maxsens_bits,
                                 float* __restrict__ out_sens) {
  int i = blockIdx.x * 256 + threadIdx.x;
  if (i < NN) out_sens[i] = raw[i] / (__uint_as_float(*maxsens_bits) + 1e-8f);
}

// ---------------------------------------------------------------------------
extern "C" void kernel_launch(void* const* d_in, const int* in_sizes, int n_in,
                              void* d_out, int out_size, void* d_ws, size_t ws_size,
                              hipStream_t stream) {
  const float* X   = (const float*)d_in[0];   // node_features (NN, NC)
  const float* INC = (const float*)d_in[1];   // incidence (NN, NE)
  const float* W1  = (const float*)d_in[2];
  const float* b1  = (const float*)d_in[3];
  const float* W2  = (const float*)d_in[4];
  const float* b2  = (const float*)d_in[5];
  const float* EA  = (const float*)d_in[6];   // edge_attention (NE,)

  float* out       = (float*)d_out;           // concat: node_out, he, attn, sens, entropy
  float* out_node  = out;
  float* out_he    = out_node + (size_t)NN * NC;
  float* out_attn  = out_he + (size_t)NE * NC;
  float* out_sens  = out_attn + NE;
  float* out_ent   = out_sens + NN;

  char* ws = (char*)d_ws;
  float*    he_part = (float*)ws;                                   // 64 * 128K floats = 32 MB
  float*    he_pre  = (float*)(ws + (size_t)KCHUNKS * NE * NC * 4); // 512 KB
  float*    degw    = (float*)((char*)he_pre + (size_t)NE * NC * 4);
  float*    raww    = degw + NN;
  unsigned* scal    = (unsigned*)(raww + NN);                       // [0]=maxdeg, [1]=maxsens

  init_scal_kernel<<<1, 32, 0, stream>>>(scal);
  edge_agg_kernel<<<dim3(NE / 128, KCHUNKS), 256, 0, stream>>>(INC, X, he_part);
  he_reduce_kernel<<<(NE * NC) / 256, 256, 0, stream>>>(he_part, he_pre);
  attn_entropy_kernel<<<1, 256, 0, stream>>>(EA, out_attn, out_ent);
  he_linear_kernel<<<(NE * NC) / 256, 256, 0, stream>>>(he_pre, W1, b1, out_attn, out_he);
  propagate_kernel<<<NN / 32, 256, 0, stream>>>(INC, out_he, W2, b2, X, out_node, degw, &scal[0]);
  sens_raw_kernel<<<NN, 128, 0, stream>>>(X, degw, &scal[0], raww, &scal[1]);
  sens_norm_kernel<<<(NN + 255) / 256, 256, 0, stream>>>(raww, &scal[1], out_sens);
}
